// GraphAttention_2_87213605912617
// MI455X (gfx1250) — matrix operand order repaired and checked by reading, not compiled, not run on hardware
//
#include <hip/hip_runtime.h>

#define Bdim 8
#define Sdim 32
#define Pdim 8
#define Vdim 8000
#define Ddim 128
#define CONCF 5.0f
#define EPSF 1e-8f

typedef __attribute__((ext_vector_type(16))) __bf16 bf16x16;
typedef __attribute__((ext_vector_type(8)))  float  f32x8;

__device__ __forceinline__ unsigned short f2bf(float x) {
  union { float f; unsigned int u; } c; c.f = x;
  unsigned int r = (c.u + 0x7FFFu + ((c.u >> 16) & 1u)) >> 16;
  return (unsigned short)r;
}
__device__ __forceinline__ unsigned int pk2bf(float lo, float hi) {
  return (unsigned int)f2bf(lo) | ((unsigned int)f2bf(hi) << 16);
}

// ---------------- K0: per-row range of edge_matrix -> 1/(rng + (rng==0)) ----
__global__ __launch_bounds__(256) void k_rowrange(const float* __restrict__ edge,
                                                  float* __restrict__ invrng) {
  __shared__ float smax[256], smin[256];
  const int row = blockIdx.x;
  const float* r = edge + (size_t)row * Vdim;
  float mx = -3.4e38f, mn = 3.4e38f;
  for (int i = threadIdx.x; i < Vdim; i += 256) {
    float v = r[i];
    mx = fmaxf(mx, v); mn = fminf(mn, v);
  }
  smax[threadIdx.x] = mx; smin[threadIdx.x] = mn;
  __syncthreads();
  for (int off = 128; off > 0; off >>= 1) {
    if (threadIdx.x < off) {
      smax[threadIdx.x] = fmaxf(smax[threadIdx.x], smax[threadIdx.x + off]);
      smin[threadIdx.x] = fminf(smin[threadIdx.x], smin[threadIdx.x + off]);
    }
    __syncthreads();
  }
  if (threadIdx.x == 0) {
    float rng = smax[0] - smin[0];
    invrng[row] = 1.0f / (rng + (rng == 0.0f ? 1.0f : 0.0f));
  }
}

// ---------------- K1: context[B,D] and 1/max(||context||, eps) --------------
__global__ __launch_bounds__(128) void k_context(
    const int* __restrict__ concepts, const int* __restrict__ clen,
    const int* __restrict__ seg_len, const float* __restrict__ embed_w,
    float* __restrict__ context, float* __restrict__ invCnorm) {
  const int b = blockIdx.x, d = threadIdx.x;
  const int seg = seg_len[b];
  float acc = 0.0f;
  for (int s = 0; s < Sdim; ++s) {
    if (s >= seg) break;                 // mask is a contiguous prefix
    int len = clen[b * Sdim + s];
    if (len > 0) {
      float m = 0.0f;
      for (int p = 0; p < len; ++p) {
        int c = concepts[(b * Sdim + s) * Pdim + p];
        m += embed_w[(size_t)c * Ddim + d];
      }
      acc += m / (float)len;
    }
  }
  acc /= (float)seg;
  context[b * Ddim + d] = acc;
  __shared__ float red[128];
  red[d] = acc * acc;
  __syncthreads();
  for (int off = 64; off > 0; off >>= 1) {
    if (d < off) red[d] += red[d + off];
    __syncthreads();
  }
  if (d == 0) invCnorm[b] = 1.0f / fmaxf(sqrtf(red[0]), EPSF);
}

// ---------------- K2: b2[b,v] = lam*|cos(b,v)|, a2[v] = (1-lam)*aff ---------
__global__ __launch_bounds__(256) void k_cos(
    const float* __restrict__ kb, const float* __restrict__ context,
    const float* __restrict__ invCnorm, const float* __restrict__ aff,
    const float* __restrict__ lam, float* __restrict__ b2,
    float* __restrict__ a2) {
  __shared__ float sctx[Bdim * Ddim];
  __shared__ float sic[Bdim];
  for (int i = threadIdx.x; i < Bdim * Ddim; i += 256) sctx[i] = context[i];
  if (threadIdx.x < Bdim) sic[threadIdx.x] = invCnorm[threadIdx.x];
  __syncthreads();
  const int wave = threadIdx.x >> 5, lane = threadIdx.x & 31;
  const int v = blockIdx.x * 8 + wave;       // grid = V/8, exact
  const float4 kv = *(const float4*)(kb + (size_t)v * Ddim + lane * 4);
  float n2 = kv.x * kv.x + kv.y * kv.y + kv.z * kv.z + kv.w * kv.w;
  float dots[Bdim];
#pragma unroll
  for (int bb = 0; bb < Bdim; ++bb) {
    const float* cx = sctx + bb * Ddim + lane * 4;
    dots[bb] = kv.x * cx[0] + kv.y * cx[1] + kv.z * cx[2] + kv.w * cx[3];
  }
#pragma unroll
  for (int off = 16; off > 0; off >>= 1) {
    n2 += __shfl_xor(n2, off, 32);
#pragma unroll
    for (int bb = 0; bb < Bdim; ++bb) dots[bb] += __shfl_xor(dots[bb], off, 32);
  }
  if (lane == 0) {
    float ik = 1.0f / fmaxf(sqrtf(n2), EPSF);
    float lv = lam[v], av = aff[v];
    a2[v] = (1.0f - lv) * av;
#pragma unroll
    for (int bb = 0; bb < Bdim; ++bb)
      b2[bb * Vdim + v] = lv * fabsf(dots[bb]) * sic[bb] * ik;
  }
}

// ---------------- K3: embed_kb -> bf16, WMMA B-fragment (32x16) layout ------
// kbt index = ((kchunk*8 + ntile)*32 + lane)*16 + q
// lane<16:  K = kchunk*32 + q,      N = ntile*16 + lane
// lane>=16: K = kchunk*32 + 16+q,   N = ntile*16 + lane-16
__global__ __launch_bounds__(256) void k_kbt(const float* __restrict__ kb,
                                             unsigned short* __restrict__ kbt) {
  const int idx = blockIdx.x * 256 + threadIdx.x;   // < V*D = 1,024,000 exact
  const int q = idx & 15;
  const int L = (idx >> 4) & 31;
  const int t = (idx >> 9) & 7;
  const int c = idx >> 12;
  const int row = c * 32 + ((L >> 4) << 4) + q;
  const int col = t * 16 + (L & 15);
  kbt[idx] = f2bf(kb[(size_t)row * Ddim + col]);
}

// ---------------- K4: fused softmax + attn @ KB via WMMA bf16 ---------------
// 64 vocab entries per iteration: 2 WMMAs per barrier, 4-deep LDS buffer ring,
// float2 streaming loads, B-fragments software-pipelined one iteration ahead.
__global__ __launch_bounds__(256) void k_attn(
    const int* __restrict__ concepts, const int* __restrict__ clen,
    const int* __restrict__ seg_len, const float* __restrict__ edge,
    const float* __restrict__ invrng, const float* __restrict__ b2,
    const float* __restrict__ a2, const unsigned short* __restrict__ kbt,
    float* __restrict__ out) {
  const int tile = blockIdx.x;        // 0..15  (two s values per tile)
  const int b = blockIdx.y;           // 0..7
  const int tid = threadIdx.x;
  const int lane = tid & 31;
  const int wave = tid >> 5;          // owns N-tile `wave` of D, rows wave/wave+8
  const int s0 = 2 * tile;
  const int r0 = wave, r1 = wave + 8; // tile rows this thread produces exps for
  const int c0 = concepts[(b * Sdim + s0) * Pdim + r0];
  const int c1 = concepts[(b * Sdim + s0 + 1) * Pdim + r0];
  const float ir0 = invrng[c0], ir1 = invrng[c1];
  const float* erow0 = edge + (size_t)c0 * Vdim;
  const float* erow1 = edge + (size_t)c1 * Vdim;
  const float* b2b = b2 + b * Vdim;

  __shared__ alignas(16) unsigned short pbuf[4][16][32];  // bf16 probs, 4-ring
  __shared__ float sden[16];
  if (tid < 16) sden[tid] = 0.0f;

  float dsum0 = 0.0f, dsum1 = 0.0f;
  f32x8 acc0 = {}, acc1 = {};
  const int m = lane & 15;            // A-fragment row this lane holds
  const int o = (lane >> 4) << 4;     // K offset: 0 (lanes<16) / 16 (lanes>=16)
  const int colLocal = (lane & 15) * 2;  // packed column pair within sub-chunk
  const int sub = lane >> 4;             // which 32-wide sub-chunk lane feeds

  // preload B fragments for iteration 0 (chunks 0 and 1)
  uint4 nb00, nb01, nb10, nb11;
  {
    const uint4* bp0 = (const uint4*)(kbt + ((size_t)(0 * 8 + wave) * 32 + lane) * 16);
    const uint4* bp1 = (const uint4*)(kbt + ((size_t)(1 * 8 + wave) * 32 + lane) * 16);
    nb00 = bp0[0]; nb01 = bp0[1]; nb10 = bp1[0]; nb11 = bp1[1];
  }

  const int NITER = Vdim / 64;        // 125
  for (int c = 0; c < NITER; ++c) {
    const int v0 = c * 64 + lane * 2; // two adjacent vocab columns per lane
    const float2 ee0 = *(const float2*)(erow0 + v0);
    const float2 ee1 = *(const float2*)(erow1 + v0);
    const float2 bb2 = *(const float2*)(b2b + v0);
    const float2 aa2 = *(const float2*)(a2 + v0);
    if (v0 + 512 < Vdim) {
      __builtin_prefetch(erow0 + v0 + 512, 0, 1);   // global_prefetch_b8
      __builtin_prefetch(erow1 + v0 + 512, 0, 1);
    }
    // w = lam*em*cos + (e>0)*(1-lam)*aff ; em = e * invrng ; p = exp(5w) <= e^5
    const float p00 = __expf(CONCF * (ee0.x * ir0 * bb2.x + (ee0.x > 0.0f ? aa2.x : 0.0f)));
    const float p01 = __expf(CONCF * (ee0.y * ir0 * bb2.y + (ee0.y > 0.0f ? aa2.y : 0.0f)));
    const float p10 = __expf(CONCF * (ee1.x * ir1 * bb2.x + (ee1.x > 0.0f ? aa2.x : 0.0f)));
    const float p11 = __expf(CONCF * (ee1.y * ir1 * bb2.y + (ee1.y > 0.0f ? aa2.y : 0.0f)));
    dsum0 += p00 + p01;
    dsum1 += p10 + p11;
    const int bufA = (c & 1) * 2;     // ring slots {bufA, bufA+1} this iteration
    const int wbuf = bufA + sub;      // this lane's columns land in sub-chunk `sub`
    *(unsigned int*)&pbuf[wbuf][r0][colLocal] = pk2bf(p00, p01);
    *(unsigned int*)&pbuf[wbuf][r1][colLocal] = pk2bf(p10, p11);
    __syncthreads();
    // A fragments (16x32 bf16): lane takes K {o..o+15} of row m
    union { uint4 u[2]; bf16x16 v; } A0, A1, B0, B1;
    A0.u[0] = *(const uint4*)&pbuf[bufA][m][o];
    A0.u[1] = *(const uint4*)&pbuf[bufA][m][o + 8];
    A1.u[0] = *(const uint4*)&pbuf[bufA + 1][m][o];
    A1.u[1] = *(const uint4*)&pbuf[bufA + 1][m][o + 8];
    B0.u[0] = nb00; B0.u[1] = nb01;
    B1.u[0] = nb10; B1.u[1] = nb11;
    if (c + 1 < NITER) {              // pipeline next iteration's B fragments
      const uint4* bp0 =
          (const uint4*)(kbt + ((size_t)((2 * c + 2) * 8 + wave) * 32 + lane) * 16);
      const uint4* bp1 =
          (const uint4*)(kbt + ((size_t)((2 * c + 3) * 8 + wave) * 32 + lane) * 16);
      nb00 = bp0[0]; nb01 = bp0[1]; nb10 = bp1[0]; nb11 = bp1[1];
    }
    acc0 = __builtin_amdgcn_wmma_f32_16x16x32_bf16(false, A0.v, false, B0.v,
                                                   (short)0, acc0, false, false);
    acc1 = __builtin_amdgcn_wmma_f32_16x16x32_bf16(false, A1.v, false, B1.v,
                                                   (short)0, acc1, false, false);
  }
  atomicAdd(&sden[r0], dsum0);   // ds_add_f32: softmax denominators
  atomicAdd(&sden[r1], dsum1);
  __syncthreads();

  // C layout: lane<16 -> rows 0..7 (s0), lane>=16 -> rows 8..15 (s0+1);
  // column N = lane%16 within this wave's D-tile. Ragged mean over p in-lane.
  const f32x8 acc = acc0 + acc1;
  const int srow = s0 + (lane >> 4);
  const int base = (lane < 16) ? 0 : 8;
  const int len = clen[b * Sdim + srow];
  const float invLen = 1.0f / (float)(len < 1 ? 1 : len);
  const float smask = (srow < seg_len[b]) ? 1.0f : 0.0f;
  float val = 0.0f;
#pragma unroll
  for (int k = 0; k < 8; ++k) {
    if (k < len) val += acc[k] * (1.0f / sden[base + k]);
  }
  val *= invLen * smask;
  out[((size_t)(b * Sdim + srow)) * Ddim + wave * 16 + (lane & 15)] = val;
}

// ---------------------------------------------------------------------------
extern "C" void kernel_launch(void* const* d_in, const int* in_sizes, int n_in,
                              void* d_out, int out_size, void* d_ws,
                              size_t ws_size, hipStream_t stream) {
  (void)in_sizes; (void)n_in; (void)out_size; (void)ws_size;
  const int*   concepts = (const int*)d_in[0];
  const int*   clen     = (const int*)d_in[1];
  const int*   seg      = (const int*)d_in[2];
  const float* embed_w  = (const float*)d_in[3];
  const float* kb       = (const float*)d_in[4];
  const float* edge     = (const float*)d_in[5];
  const float* aff      = (const float*)d_in[6];
  const float* lam      = (const float*)d_in[7];
  float* out = (float*)d_out;

  char* ws = (char*)d_ws;
  unsigned short* kbt = (unsigned short*)ws; ws += (size_t)Vdim * Ddim * 2; // 2.0 MB
  float* invrng   = (float*)ws; ws += (size_t)Vdim * 4;
  float* context  = (float*)ws; ws += (size_t)Bdim * Ddim * 4;
  float* invCnorm = (float*)ws; ws += (size_t)Bdim * 4;
  float* b2       = (float*)ws; ws += (size_t)Bdim * Vdim * 4;
  float* a2       = (float*)ws; ws += (size_t)Vdim * 4;

  k_rowrange<<<Vdim, 256, 0, stream>>>(edge, invrng);
  k_context<<<Bdim, 128, 0, stream>>>(concepts, clen, seg, embed_w, context,
                                      invCnorm);
  k_cos<<<Vdim / 8, 256, 0, stream>>>(kb, context, invCnorm, aff, lam, b2, a2);
  k_kbt<<<(Vdim * Ddim) / 256, 256, 0, stream>>>(kb, kbt);
  k_attn<<<dim3(Sdim / 2, Bdim), 256, 0, stream>>>(concepts, clen, seg, edge,
                                                   invrng, b2, a2, kbt, out);
}
